// RUNG_combined_model_39505109188722
// MI455X (gfx1250) — compile-verified, matrix operand order checked by reading, not run on hardware
//
#include <hip/hip_runtime.h>
#include <hip/hip_bf16.h>
#include <stdint.h>

typedef __attribute__((ext_vector_type(16))) _Float16 v16h;
typedef __attribute__((ext_vector_type(8)))  _Float16 v8h;
typedef __attribute__((ext_vector_type(8)))  float    v8f;

#define NN    4096
#define OUTD  64
#define NBINS 8192
#define CAP   2048

__device__ __forceinline__ float scad_w(float y, float lamc) {
    const float a = 3.7f;
    if (y <= lamc) return 1.0f;
    if (y <= a * lamc) return (a * lamc - y) / ((a - 1.0f) * fmaxf(y, 1e-8f));
    return 0.0f;
}

// Async global->LDS copy (CDNA5 GLOBAL_LOAD_ASYNC_TO_LDS_B32, ASYNCcnt-tracked).
// Low 32 bits of a generic shared pointer are the LDS byte address.
__device__ __forceinline__ void async_g2l_b32(const float* gptr, float* lptr) {
    uint32_t loff = (uint32_t)(uintptr_t)lptr;
    asm volatile("global_load_async_to_lds_b32 %0, %1, off"
                 :: "v"(loff), "v"(gptr) : "memory");
}
__device__ __forceinline__ void wait_async0() {
    asm volatile("s_wait_asynccnt 0x0" ::: "memory");
}

// ---------------- degree + edge count ----------------
__global__ void k_degree(const float* __restrict__ A, float* __restrict__ D,
                         int* __restrict__ edgeTotal) {
    int r = blockIdx.x, t = threadIdx.x;
    float s = 0.0f; int c = 0;
    const float* Arow = A + (size_t)r * NN;
    for (int j = t; j < NN; j += 256) {
        float a = Arow[j];
        s += a;
        if (a > 0.0f) c++;
    }
    __shared__ float ss[256];
    __shared__ int   sc[256];
    ss[t] = s; sc[t] = c;
    __syncthreads();
    for (int o = 128; o > 0; o >>= 1) {
        if (t < o) { ss[t] += ss[t + o]; sc[t] += sc[t + o]; }
        __syncthreads();
    }
    if (t == 0) { D[r] = 1.0f + ss[0]; atomicAdd(edgeTotal, sc[0]); }
}

__global__ void k_zero_i32(int* __restrict__ p, int n) {
    int i = blockIdx.x * 256 + threadIdx.x;
    if (i < n) p[i] = 0;
}

// ---------------- f32 -> f16 pack / transpose-pack ----------------
__global__ void k_pack_f16(const float* __restrict__ in, _Float16* __restrict__ out, int n) {
    int i = blockIdx.x * 256 + threadIdx.x;
    if (i < n) out[i] = (_Float16)in[i];
}
// out[n*K + k] = in[k*N + n]   (pack B into B^T, f16)
__global__ void k_transpose_f16(const float* __restrict__ in, _Float16* __restrict__ out,
                                int K, int N) {
    int i = blockIdx.x * 256 + threadIdx.x;
    if (i < K * N) {
        int k = i / N, n = i % N;
        out[(size_t)n * K + k] = (_Float16)in[i];
    }
}

// ---------------- WMMA GEMM: C = act(A[MxK] @ Bt[NxK]^T + bias) ----------------
// A, Bt pre-packed f16 row-major. One 16x16 tile per wave32, K fully unrolled.
template <int K, bool RELU, bool OUT_F16, bool DUAL>
__global__ void k_gemm_wmma(const _Float16* __restrict__ A, const _Float16* __restrict__ Bt,
                            const float* __restrict__ bias, void* __restrict__ C0,
                            float* __restrict__ C2, int N) {
    int wave = threadIdx.x >> 5;
    int lane = threadIdx.x & 31;
    int tile = blockIdx.x * 8 + wave;
    int tiles_n = N >> 4;
    int tm = tile / tiles_n, tn = tile % tiles_n;

    int arow = tm * 16 + (lane & 15);   // A frag: lane%16 = M row
    int brow = tn * 16 + (lane & 15);   // B frag: lane%16 = N col (rows of Bt)
    int kb   = (lane >> 4) * 8;         // lane half selects K sub-block

    const _Float16* ap = A  + (size_t)arow * K + kb;
    const _Float16* bp = Bt + (size_t)brow * K + kb;

    v8f acc = {};
#pragma unroll
    for (int k0 = 0; k0 < K; k0 += 32) {
        v8h alo = *(const v8h*)(ap + k0);
        v8h ahi = *(const v8h*)(ap + k0 + 16);
        v8h blo = *(const v8h*)(bp + k0);
        v8h bhi = *(const v8h*)(bp + k0 + 16);
        v16h af = __builtin_shufflevector(alo, ahi, 0,1,2,3,4,5,6,7,8,9,10,11,12,13,14,15);
        v16h bf = __builtin_shufflevector(blo, bhi, 0,1,2,3,4,5,6,7,8,9,10,11,12,13,14,15);
        acc = __builtin_amdgcn_wmma_f32_16x16x32_f16(false, af, false, bf,
                                                     (short)0, acc, false, false);
    }
    // C/D layout: lane%16 = N, VGPR v -> M = (lane/16)*8 + v
    int n  = tn * 16 + (lane & 15);
    int mb = tm * 16 + (lane >> 4) * 8;
    float bv = bias[n];
#pragma unroll
    for (int v = 0; v < 8; v++) {
        float val = acc[v] + bv;
        if (RELU) val = fmaxf(val, 0.0f);
        if (OUT_F16) ((_Float16*)C0)[(size_t)(mb + v) * N + n] = (_Float16)val;
        else         ((float*)C0)[(size_t)(mb + v) * N + n] = val;
        if (DUAL)    C2[(size_t)(mb + v) * N + n] = val;
    }
}

// ---------------- row-normalize: Fu = normalize(Fc / sqrt(D)) ----------------
__global__ void k_normalize(const float* __restrict__ Fc, const float* __restrict__ D,
                            float* __restrict__ Fu) {
    int r = blockIdx.x, lane = threadIdx.x;   // 32 threads
    float ds = sqrtf(D[r]);
    float f0 = Fc[(size_t)r * OUTD + lane]      / ds;
    float f1 = Fc[(size_t)r * OUTD + lane + 32] / ds;
    float ss = f0 * f0 + f1 * f1;
    for (int o = 16; o > 0; o >>= 1) ss += __shfl_xor(ss, o, 32);
    float inv = 1.0f / fmaxf(sqrtf(ss), 1e-8f);
    Fu[(size_t)r * OUTD + lane]      = f0 * inv;
    Fu[(size_t)r * OUTD + lane + 32] = f1 * inv;
}

// ---------------- histogram of edge cosine distances ----------------
__global__ void k_hist(const float* __restrict__ A, const float* __restrict__ Fu,
                       int* __restrict__ hist) {
    int r = blockIdx.x, lane = threadIdx.x;   // 32 threads
    __shared__ float fuI[OUTD];
    async_g2l_b32(Fu + (size_t)r * OUTD + lane,      &fuI[lane]);
    async_g2l_b32(Fu + (size_t)r * OUTD + lane + 32, &fuI[lane + 32]);
    wait_async0();
    __syncthreads();
    const float* Arow = A + (size_t)r * NN;
    for (int base = 0; base < NN; base += 32) {
        __builtin_prefetch(Arow + base + 2048, 0, 1);
        int j = base + lane;
        float a = Arow[j];
        if (a > 0.0f && j != r) {
            const float* fj = Fu + (size_t)j * OUTD;
            float dot = 0.0f;
#pragma unroll 8
            for (int d = 0; d < OUTD; d++) dot += fuI[d] * fj[d];
            float y = fminf(fmaxf(1.0f - dot, 0.0f), 2.0f);
            int b = (int)(y * (NBINS * 0.5f));
            b = b < (NBINS - 1) ? b : (NBINS - 1);
            atomicAdd(&hist[b], 1);
        }
    }
}

// ---------------- quantile + scalar threshold ----------------
__global__ void k_quantile(const int* __restrict__ hist, const int* __restrict__ edgeTotal,
                           const float* __restrict__ lg0, const float* __restrict__ rdec,
                           const float* __restrict__ ralpha, int step,
                           float* __restrict__ lamOut) {
    if (threadIdx.x != 0) return;
    const float SCAD_A = 3.7f;
    int cnt = *edgeTotal;
    float gd = 1.0f;
    if (cnt > 0) {
        float kq = 0.75f * (float)(cnt - 1);
        int kf = (int)kq;
        float frac = kq - (float)kf;
        float v0 = 2.0f, v1 = 2.0f;
        long cum = 0; bool got0 = false, got1 = false;
        for (int b = 0; b < NBINS; b++) {
            int h = hist[b];
            if (!h) continue;
            long lo = cum; cum += h;
            if (!got0 && (long)kf < cum) {
                v0 = ((float)b + ((float)(kf - lo) + 0.5f) / (float)h) * (2.0f / NBINS);
                got0 = true;
            }
            if (!got1 && (long)(kf + 1) < cum) {
                v1 = ((float)b + ((float)(kf + 1 - lo) + 0.5f) / (float)h) * (2.0f / NBINS);
                got1 = true;
                break;
            }
        }
        if (!got1) v1 = v0;
        gd = v0 * (1.0f - frac) + v1 * frac;
    }
    gd = fmaxf(gd, 1e-8f);
    float g0 = expf(lg0[0]);
    float rr = 1.0f / (1.0f + expf(-rdec[0]));
    float al = 1.0f / (1.0f + expf(-ralpha[0]));
    float gp = g0 * powf(rr, (float)step);
    lamOut[0] = al * (gp / SCAD_A) + (1.0f - al) * (gd / SCAD_A);
}

// ---------------- sparse propagate: one wave per row, ballot-compacted ----------------
__global__ void k_prop(const float* __restrict__ A, const float* __restrict__ Fu,
                       const float* __restrict__ Fcin, const float* __restrict__ F0,
                       const float* __restrict__ D, const float* __restrict__ lamC,
                       float* __restrict__ Fcout) {
    const float LAM = 1.0f / 0.9f - 1.0f;
    int r = blockIdx.x, lane = threadIdx.x;   // 32 threads
    __shared__ float fuI[OUTD];
    __shared__ int   jl[CAP];
    __shared__ float sl[CAP];
    async_g2l_b32(Fu + (size_t)r * OUTD + lane,      &fuI[lane]);
    async_g2l_b32(Fu + (size_t)r * OUTD + lane + 32, &fuI[lane + 32]);
    wait_async0();
    __syncthreads();
    float lamc = lamC[0];
    float Dr   = D[r];
    float isr  = rsqrtf(Dr);
    float acc0 = 0.0f, acc1 = 0.0f, qp = 0.0f;
    int cnt = 0;
    const float* Arow = A + (size_t)r * NN;
    for (int base = 0; base < NN; base += 32) {
        __builtin_prefetch(Arow + base + 2048, 0, 1);
        int j = base + lane;
        float a = Arow[j];
        bool act = (a > 0.0f) && (j != r);
        float s = 0.0f;
        if (act) {
            const float* fj = Fu + (size_t)j * OUTD;
            float dot = 0.0f;
#pragma unroll 8
            for (int d = 0; d < OUTD; d++) dot += fuI[d] * fj[d];
            float y = fminf(fmaxf(1.0f - dot, 0.0f), 2.0f);
            float w = scad_w(y, lamc);
            qp += w;
            s = w * isr * rsqrtf(D[j]);   // w * A_tilde factor
        }
        unsigned mask = (unsigned)__ballot(act);
        int pos = __popc(mask & ((1u << lane) - 1u));
        if (act) { jl[cnt + pos] = j; sl[cnt + pos] = s; }
        cnt += __popc(mask);               // wave-uniform
        if (cnt > CAP - 32) {
            __syncthreads();
            for (int e = 0; e < cnt; e++) {
                float sv = sl[e];
                const float* fr = Fcin + (size_t)jl[e] * OUTD;
                acc0 += sv * fr[lane];
                acc1 += sv * fr[lane + 32];
            }
            __syncthreads();
            cnt = 0;
        }
    }
    __syncthreads();
    for (int e = 0; e < cnt; e++) {
        float sv = sl[e];
        const float* fr = Fcin + (size_t)jl[e] * OUTD;
        acc0 += sv * fr[lane];
        acc1 += sv * fr[lane + 32];
    }
    for (int o = 16; o > 0; o >>= 1) qp += __shfl_xor(qp, o, 32);
    float qhat = qp / Dr + LAM;
    float inv = 1.0f / qhat;
    Fcout[(size_t)r * OUTD + lane]      = (acc0 + LAM * F0[(size_t)r * OUTD + lane])      * inv;
    Fcout[(size_t)r * OUTD + lane + 32] = (acc1 + LAM * F0[(size_t)r * OUTD + lane + 32]) * inv;
}

extern "C" void kernel_launch(void* const* d_in, const int* in_sizes, int n_in,
                              void* d_out, int out_size, void* d_ws, size_t ws_size,
                              hipStream_t stream) {
    (void)in_sizes; (void)n_in; (void)out_size; (void)ws_size;
    const float* A      = (const float*)d_in[0];
    const float* X      = (const float*)d_in[1];
    const float* W1     = (const float*)d_in[2];
    const float* b1     = (const float*)d_in[3];
    const float* W2     = (const float*)d_in[4];
    const float* b2     = (const float*)d_in[5];
    const float* lg0    = (const float*)d_in[6];
    const float* rdec   = (const float*)d_in[7];
    const float* ralpha = (const float*)d_in[8];
    float* out = (float*)d_out;

    // workspace layout (floats; all partitions 16B-aligned)
    float* ws   = (float*)d_ws;
    float* D    = ws;                        // 4096
    float* F0   = D + NN;                    // 4096*64
    float* FcA  = F0 + (size_t)NN * OUTD;
    float* FcB  = FcA + (size_t)NN * OUTD;
    float* Fu   = FcB + (size_t)NN * OUTD;
    float* lamC = Fu + (size_t)NN * OUTD;    // 4 (padded)
    int*   hist = (int*)(lamC + 4);          // NBINS
    int*   edgeTotal = hist + NBINS;         // 4 (padded)
    _Float16* Xh   = (_Float16*)(edgeTotal + 4);        // 4096*512
    _Float16* W1t  = Xh + (size_t)NN * 512;              // 256*512
    _Float16* W2t  = W1t + (size_t)256 * 512;            // 64*256
    _Float16* H1h  = W2t + (size_t)64 * 256;             // 4096*256

    // zero hist + edgeTotal, degree/edge-count
    k_zero_i32<<<(NBINS + 4 + 255) / 256, 256, 0, stream>>>(hist, NBINS + 4);
    k_degree<<<NN, 256, 0, stream>>>(A, D, edgeTotal);

    // pack f16 operands
    k_pack_f16<<<(NN * 512 + 255) / 256, 256, 0, stream>>>(X, Xh, NN * 512);
    k_transpose_f16<<<(512 * 256 + 255) / 256, 256, 0, stream>>>(W1, W1t, 512, 256);
    k_transpose_f16<<<(256 * 64 + 255) / 256, 256, 0, stream>>>(W2, W2t, 256, 64);

    // MLP via WMMA: H1h = relu(X@W1+b1) [f16]; F0 = H1@W2+b2 (also seeds FcA)
    k_gemm_wmma<512, true, true, false>
        <<<(NN / 16) * (256 / 16) / 8, 256, 0, stream>>>(Xh, W1t, b1, (void*)H1h, nullptr, 256);
    k_gemm_wmma<256, false, false, true>
        <<<(NN / 16) * (OUTD / 16) / 8, 256, 0, stream>>>(H1h, W2t, b2, (void*)F0, FcA, OUTD);

    const float* fin = FcA;
    for (int k = 0; k < 4; k++) {
        float* fout = (k == 3) ? out : ((k & 1) ? FcA : FcB);
        k_normalize<<<NN, 32, 0, stream>>>(fin, D, Fu);
        k_zero_i32<<<(NBINS + 255) / 256, 256, 0, stream>>>(hist, NBINS);
        k_hist<<<NN, 32, 0, stream>>>(A, Fu, hist);
        k_quantile<<<1, 32, 0, stream>>>(hist, edgeTotal, lg0, rdec, ralpha, k, lamC);
        k_prop<<<NN, 32, 0, stream>>>(A, Fu, fin, F0, D, lamC, fout);
        fin = fout;
    }
}
